// HSTUModel_83794811945340
// MI455X (gfx1250) — matrix operand order repaired and checked by reading, not compiled, not run on hardware
//
#include <hip/hip_runtime.h>

#define S_LEN 2048
#define HID   1024
#define NHEAD 8
#define LDIM  64
#define ADIM  64
#define NBUCK 128
#define BATCH 4
#define DPROJ 2048   // 2*L*NH + 2*A*NH

typedef __attribute__((ext_vector_type(16))) __bf16       v16bf;
typedef __attribute__((ext_vector_type(8)))  float        v8f;
typedef __attribute__((ext_vector_type(4)))  unsigned int u32x4;
typedef __attribute__((ext_vector_type(4)))  float        f32x4;
typedef __attribute__((ext_vector_type(2)))  float        f32x2;

union FragU { u32x4 q[2]; v16bf v; };

__device__ __forceinline__ v16bf load_frag2(const unsigned short* p0, const unsigned short* p1) {
  FragU f;
  f.q[0] = *(const u32x4*)p0;
  f.q[1] = *(const u32x4*)p1;
  return f.v;
}
// A-operand fragment from LDS in chunked layout: half-addr = ((chunk*32 + row) * 8)
// chunk c holds cols [8c, 8c+8) of each row; fragment needs chunks (kk/8 + h) and (kk/8 + 2 + h)
__device__ __forceinline__ v16bf frag_a_ch(const unsigned short* sBase, int kk, int row, int hsel) {
  int c0 = (kk >> 3) + hsel;
  int c1 = c0 + 2;
  return load_frag2(sBase + (((c0 << 5) + row) << 3), sBase + (((c1 << 5) + row) << 3));
}
// A-operand fragment, plain row layout
__device__ __forceinline__ v16bf frag_a_p(const unsigned short* base, int hsel) {
  return load_frag2(base + 8 * hsel, base + 16 + 8 * hsel);
}
// B-operand (32x16 bf16): lane n = lane&15; halfs [8h,8h+8) and [16+8h,24+8h)
__device__ __forceinline__ v16bf frag_b(const unsigned short* base, int hsel) {
  return load_frag2(base + 8 * hsel, base + 16 + 8 * hsel);
}
__device__ __forceinline__ v8f wmma_bf16(v16bf a, v16bf b, v8f c) {
  return __builtin_amdgcn_wmma_f32_16x16x32_bf16(false, a, false, b, (short)0, c, false, false);
}
// native f32 -> bf16 convert (single HW instruction, fusable into packed form)
__device__ __forceinline__ unsigned short f2bf(float f) {
  union { __bf16 h; unsigned short u; } cv;
  cv.h = (__bf16)f;
  return cv.u;
}
__device__ __forceinline__ unsigned int pack_bf2(float lo, float hi) {
  return (unsigned int)f2bf(lo) | ((unsigned int)f2bf(hi) << 16);
}
// silu via hardware v_rcp_f32 (avoid IEEE division Newton chains in hot loops)
__device__ __forceinline__ float silu_f(float x) {
  return x * __builtin_amdgcn_rcpf(1.0f + __expf(-x));
}
// fast tanh: 1 - 2/(exp(2x)+1)
__device__ __forceinline__ float tanh_f(float x) {
  return 1.0f - 2.0f * __builtin_amdgcn_rcpf(__expf(2.0f * x) + 1.0f);
}

// ---------------------------------------------------------------- weight prep
__global__ void transpose_bf16_kernel(const float* __restrict__ src,
                                      unsigned short* __restrict__ dst, int K, int N) {
  int idx = blockIdx.x * blockDim.x + threadIdx.x;
  if (idx >= K * N) return;
  int k = idx / N, n = idx - k * N;
  dst[n * K + k] = f2bf(src[idx]);   // src[K][N] -> dst[N][K]
}

// FiLM has only 4 possible ids -> precompute rb table [4][2*HID]
__global__ void film_table_kernel(const float* __restrict__ act_tab, const float* __restrict__ g,
                                  const float* __restrict__ bb, const float* __restrict__ film_w,
                                  const float* __restrict__ film_b, float* __restrict__ rb_tab) {
  int id = blockIdx.x;
  const float* a = act_tab + id * 32;
  float m = 0.f;
  for (int k = 0; k < 32; k++) m += a[k];
  m *= (1.f / 32.f);
  float v = 0.f;
  for (int k = 0; k < 32; k++) { float d = a[k] - m; v += d * d; }
  v *= (1.f / 32.f);
  float inv = rsqrtf(v + 1e-5f);
  float lnae[32];
  for (int k = 0; k < 32; k++) lnae[k] = (a[k] - m) * inv * g[k] + bb[k];
  for (int j = threadIdx.x; j < 2 * HID; j += blockDim.x) {
    float s = film_b[j];
    for (int k = 0; k < 32; k++) s += lnae[k] * film_w[k * 2 * HID + j];
    rb_tab[id * 2 * HID + j] = s;
  }
}

// -------------------------------------------- fused LN + proj GEMM + silu + RoPE
__global__ __launch_bounds__(256) void ln_proj_kernel(
    const float* __restrict__ x, const float* __restrict__ ln_g, const float* __restrict__ ln_b,
    const unsigned short* __restrict__ wT,   // [DPROJ][HID] bf16 (pre-transposed)
    float* __restrict__ Ubuf,                // [B,NH,S,L] f32
    unsigned short* __restrict__ Vt,         // [B,NH,L,S] bf16 (transposed for B-operand)
    unsigned short* __restrict__ Qb,         // [B,NH,S,A] bf16 (roped)
    unsigned short* __restrict__ Kb) {       // [B,NH,S,A] bf16 (roped)
  // chunked LDS: 128 chunks (HID/8) x 32 rows x 8 halfs = 64KB, conflict-free frag reads
  __shared__ unsigned short sH[32 * HID];
  unsigned int* sH32 = (unsigned int*)sH;
  const int tid = threadIdx.x;
  const int w = tid >> 5, lane = tid & 31;
  const int m0 = blockIdx.x * 32;            // global row base (b*S + s)
  const int bI = m0 >> 11;
  const int sbase = m0 & (S_LEN - 1);

  // phase 1: LayerNorm 4 rows per wave (float4 loads, packed bf16 LDS stores)
  for (int rr = 0; rr < 4; rr++) {
    int row = w * 4 + rr;
    const f32x4* xr4 = (const f32x4*)(x + (size_t)(m0 + row) * HID);
    f32x4 vv[8];
    float s = 0.f, s2 = 0.f;
#pragma unroll
    for (int i = 0; i < 8; i++) {
      f32x4 t = xr4[lane + 32 * i];
      vv[i] = t;
      s += t.x + t.y + t.z + t.w;
      s2 += t.x * t.x + t.y * t.y + t.z * t.z + t.w * t.w;
    }
#pragma unroll
    for (int msk = 16; msk >= 1; msk >>= 1) { s += __shfl_xor(s, msk); s2 += __shfl_xor(s2, msk); }
    float mean = s * (1.f / HID);
    float inv = rsqrtf(s2 * (1.f / HID) - mean * mean + 1e-5f);
#pragma unroll
    for (int i = 0; i < 8; i++) {
      int c0 = 4 * lane + 128 * i;
      float h0 = (vv[i].x - mean) * inv * ln_g[c0 + 0] + ln_b[c0 + 0];
      float h1 = (vv[i].y - mean) * inv * ln_g[c0 + 1] + ln_b[c0 + 1];
      float h2 = (vv[i].z - mean) * inv * ln_g[c0 + 2] + ln_b[c0 + 2];
      float h3 = (vv[i].w - mean) * inv * ln_g[c0 + 3] + ln_b[c0 + 3];
      int chunk = c0 >> 3, off = c0 & 7;
      int u32i = (((chunk << 5) + row) << 3 | off) >> 1;
      sH32[u32i]     = pack_bf2(h0, h1);
      sH32[u32i + 1] = pack_bf2(h2, h3);
    }
  }
  __syncthreads();

  // phase 2: each wave computes a 32x64 slab; rolled K-loop (no spill-inducing hoist)
  const int mlane = lane & 15, hsel = lane >> 4;

  for (int it = 0; it < 4; it++) {
    int n0 = it * 512 + w * 64;              // 64-col block, head- and region-aligned
    const unsigned short* bCol0 = wT + (size_t)(n0 + 0 * 16 + mlane) * HID;
    const unsigned short* bCol1 = wT + (size_t)(n0 + 1 * 16 + mlane) * HID;
    const unsigned short* bCol2 = wT + (size_t)(n0 + 2 * 16 + mlane) * HID;
    const unsigned short* bCol3 = wT + (size_t)(n0 + 3 * 16 + mlane) * HID;
    v8f acc[2][4] = {};
#pragma clang loop unroll(disable)
    for (int kk = 0; kk < HID; kk += 32) {
      v16bf a0 = frag_a_ch(sH, kk, mlane, hsel);
      v16bf a1 = frag_a_ch(sH, kk, 16 + mlane, hsel);
      v16bf b0 = frag_b(bCol0 + kk, hsel);
      acc[0][0] = wmma_bf16(a0, b0, acc[0][0]);
      acc[1][0] = wmma_bf16(a1, b0, acc[1][0]);
      v16bf b1 = frag_b(bCol1 + kk, hsel);
      acc[0][1] = wmma_bf16(a0, b1, acc[0][1]);
      acc[1][1] = wmma_bf16(a1, b1, acc[1][1]);
      v16bf b2 = frag_b(bCol2 + kk, hsel);
      acc[0][2] = wmma_bf16(a0, b2, acc[0][2]);
      acc[1][2] = wmma_bf16(a1, b2, acc[1][2]);
      v16bf b3 = frag_b(bCol3 + kk, hsel);
      acc[0][3] = wmma_bf16(a0, b3, acc[0][3]);
      acc[1][3] = wmma_bf16(a1, b3, acc[1][3]);
    }
    int region = n0 >> 9;                    // 0:U 1:V 2:Q 3:K (uniform per wave)
    int h = (n0 & 511) >> 6;
#pragma unroll
    for (int nc = 0; nc < 4; nc++) {
      int cl = nc * 16 + mlane;              // 0..63 within head
      // RoPE frequency depends only on the column pair -> hoist out of tile/r loops
      float freq = __expf(-(float)(cl >> 1) * (9.210340371976184f / 32.0f)); // theta^(-2c/A)
#pragma unroll
      for (int tile = 0; tile < 2; tile++) {
#pragma unroll
        for (int r = 0; r < 8; r++) {
          float v = silu_f(acc[tile][nc][r]);
          float pv = __shfl_xor(v, 1);       // RoPE partner (adjacent column)
          int srow = sbase + tile * 16 + r + 8 * hsel;
          if (region == 0) {
            Ubuf[(((size_t)(bI * NHEAD + h)) * S_LEN + srow) * LDIM + cl] = v;
          } else if (region == 1) {
            Vt[(((size_t)(bI * NHEAD + h)) * LDIM + cl) * S_LEN + srow] = f2bf(v);
          } else {
            unsigned short* dst = (region == 2) ? Qb : Kb;
            float f = (float)srow * freq;
            float cs = __cosf(f), sn = __sinf(f);
            float o = ((cl & 1) == 0) ? (v * cs - pv * sn) : (v * cs + pv * sn);
            dst[(((size_t)(bI * NHEAD + h)) * S_LEN + srow) * ADIM + cl] = f2bf(o);
          }
        }
      }
    }
  }
}

// ------------------------------- fused causal attention (no softmax -> streaming)
// Block = (batch, 16-query-row tile), 8 waves = 8 heads sharing one bias tile.
#define SSTRIDE 40   // padded score-row stride (halfs): 16 lanes hit distinct banks
__global__ __launch_bounds__(256) void attn_kernel(
    const unsigned short* __restrict__ Qb, const unsigned short* __restrict__ Kb,
    const unsigned short* __restrict__ Vt, const float* __restrict__ Ubuf,
    const int* __restrict__ interval, const float* __restrict__ ts_w,
    const float* __restrict__ pos_w, unsigned short* __restrict__ ud) {
  __shared__ float sBias[2][16 * 32];             // double-buffered shared bias tile
  __shared__ unsigned short sS[8][16 * SSTRIDE];  // wave-private score tiles (bf16)
  const int tid = threadIdx.x;
  const int w = tid >> 5, lane = tid & 31;
  const int mlane = lane & 15, hsel = lane >> 4;
  const int qt = blockIdx.x, bI = blockIdx.y;
  const int q0 = qt * 16;
  const int hh = w;                               // wave == head

  const unsigned short* Qp = Qb + ((size_t)(bI * NHEAD + hh)) * S_LEN * ADIM;
  const unsigned short* Kp = Kb + ((size_t)(bI * NHEAD + hh)) * S_LEN * ADIM;
  const unsigned short* Vp = Vt + ((size_t)(bI * NHEAD + hh)) * LDIM * S_LEN;

  const unsigned short* qrow = Qp + (size_t)(q0 + mlane) * ADIM;
  v16bf aQ0 = frag_a_p(qrow, hsel);
  v16bf aQ1 = frag_a_p(qrow + 32, hsel);

  // bias-worker mapping: thread computes elements (brow, bcol) and (brow, bcol+1)
  const int brow = tid >> 4;               // 0..15
  const int bcol = (tid & 15) * 2;         // 0..30 even
  const int srow_th = q0 + brow;
  int nx = srow_th + 1; if (nx > S_LEN - 1) nx = S_LEN - 1;
  const float inext_th = (float)interval[bI * S_LEN + nx];
  const int relbase = (S_LEN - 1) - srow_th;

  v8f acc[4] = {};
  unsigned short* sw = &sS[w][0];

  const int tEnd = q0 + 16;
  int pb = 0;
  for (int t0 = 0; t0 < tEnd; t0 += 32, pb ^= 1) {
    {   // cooperative 16x32 bias tile: bias(s,t) = ts_w[bucket] + pos_w[t-s+S-1]
      int t = t0 + bcol;
      float itv0 = (float)interval[bI * S_LEN + t];
      float itv1 = (float)interval[bI * S_LEN + t + 1];
      float d0 = fmaxf(fabsf(inext_th - itv0), 1.0f);
      float d1 = fmaxf(fabsf(inext_th - itv1), 1.0f);
      int k0 = (int)(__logf(d0) * (1.0f / 0.301f));
      int k1 = (int)(__logf(d1) * (1.0f / 0.301f));
      k0 = k0 < 0 ? 0 : (k0 > NBUCK ? NBUCK : k0);
      k1 = k1 < 0 ? 0 : (k1 > NBUCK ? NBUCK : k1);
      f32x2 bv;
      bv.x = ts_w[k0] + pos_w[t + relbase];
      bv.y = ts_w[k1] + pos_w[t + 1 + relbase];
      *(f32x2*)&sBias[pb][brow * 32 + bcol] = bv;
    }
    __syncthreads();   // one barrier/step; double buffer covers cross-step WAR

#pragma unroll
    for (int sub = 0; sub < 2; sub++) {
      int tt = t0 + sub * 16;
      const unsigned short* krow = Kp + (size_t)(tt + mlane) * ADIM;
      v16bf bK0 = frag_b(krow, hsel);
      v16bf bK1 = frag_b(krow + 32, hsel);
      v8f sc = {};
      sc = wmma_bf16(aQ0, bK0, sc);
      sc = wmma_bf16(aQ1, bK1, sc);
      int t = tt + mlane;
#pragma unroll
      for (int r = 0; r < 8; r++) {
        int srow = q0 + r + 8 * hsel;
        float val = sc[r] + sBias[pb][(r + 8 * hsel) * 32 + sub * 16 + mlane];
        val = silu_f(val) * (1.0f / (float)S_LEN);
        if (t > srow) val = 0.0f;                 // causal mask
        sw[(r + 8 * hsel) * SSTRIDE + sub * 16 + mlane] = f2bf(val);
      }
    }
    v16bf aS = frag_a_p(sw + mlane * SSTRIDE, hsel);  // scores as A-operand
#pragma unroll
    for (int nc = 0; nc < 4; nc++) {
      const unsigned short* vrow = Vp + (size_t)(nc * 16 + mlane) * S_LEN + t0;
      v16bf bV = frag_b(vrow, hsel);
      acc[nc] = wmma_bf16(aS, bV, acc[nc]);
    }
  }

  // row-LN (no affine) over L=64, gate with U, store u_dot (B,S,NH*L) bf16
#pragma unroll
  for (int r = 0; r < 8; r++) {
    float s = 0.f, s2 = 0.f;
#pragma unroll
    for (int nc = 0; nc < 4; nc++) { float t = acc[nc][r]; s += t; s2 += t * t; }
#pragma unroll
    for (int msk = 8; msk >= 1; msk >>= 1) { s += __shfl_xor(s, msk); s2 += __shfl_xor(s2, msk); }
    float mean = s * (1.0f / LDIM);
    float inv = rsqrtf(s2 * (1.0f / LDIM) - mean * mean + 1e-5f);
    int srow = q0 + r + 8 * hsel;
#pragma unroll
    for (int nc = 0; nc < 4; nc++) {
      int l = nc * 16 + mlane;
      float u = Ubuf[(((size_t)(bI * NHEAD + hh)) * S_LEN + srow) * LDIM + l];
      float o = (acc[nc][r] - mean) * inv * u;
      ud[((size_t)(bI * S_LEN + srow)) * (NHEAD * LDIM) + hh * LDIM + l] = f2bf(o);
    }
  }
}

// ---------------------------------------------- output GEMM: y = x + u_dot@o_w + o_b
__global__ __launch_bounds__(256) void out_gemm_kernel(
    const unsigned short* __restrict__ ud,   // [B*S][512] bf16
    const unsigned short* __restrict__ owT,  // [HID][512] bf16 (pre-transposed)
    const float* __restrict__ x, const float* __restrict__ o_b,
    float* __restrict__ y) {
  __shared__ unsigned short sA[32 * 512];    // chunked: 64 chunks x 32 rows x 8 halfs = 32KB
  const int tid = threadIdx.x;
  const int m0 = blockIdx.x * 32;
  {
    const u32x4* src = (const u32x4*)(ud + (size_t)m0 * 512);
    u32x4* dst = (u32x4*)sA;
    for (int e = tid; e < 64 * 32; e += 256) {
      int row = e >> 6, chunk = e & 63;      // contiguous global reads
      dst[(chunk << 5) + row] = src[e];
    }
  }
  __syncthreads();

  const int w = tid >> 5, lane = tid & 31;
  const int mlane = lane & 15, hsel = lane >> 4;
  for (int it = 0; it < 2; it++) {
    int n0 = it * 512 + w * 64;
    const unsigned short* bCol0 = owT + (size_t)(n0 + 0 * 16 + mlane) * 512;
    const unsigned short* bCol1 = owT + (size_t)(n0 + 1 * 16 + mlane) * 512;
    const unsigned short* bCol2 = owT + (size_t)(n0 + 2 * 16 + mlane) * 512;
    const unsigned short* bCol3 = owT + (size_t)(n0 + 3 * 16 + mlane) * 512;
    v8f acc[2][4] = {};
#pragma clang loop unroll(disable)
    for (int kk = 0; kk < 512; kk += 32) {
      v16bf a0 = frag_a_ch(sA, kk, mlane, hsel);
      v16bf a1 = frag_a_ch(sA, kk, 16 + mlane, hsel);
      v16bf b0 = frag_b(bCol0 + kk, hsel);
      acc[0][0] = wmma_bf16(a0, b0, acc[0][0]);
      acc[1][0] = wmma_bf16(a1, b0, acc[1][0]);
      v16bf b1 = frag_b(bCol1 + kk, hsel);
      acc[0][1] = wmma_bf16(a0, b1, acc[0][1]);
      acc[1][1] = wmma_bf16(a1, b1, acc[1][1]);
      v16bf b2 = frag_b(bCol2 + kk, hsel);
      acc[0][2] = wmma_bf16(a0, b2, acc[0][2]);
      acc[1][2] = wmma_bf16(a1, b2, acc[1][2]);
      v16bf b3 = frag_b(bCol3 + kk, hsel);
      acc[0][3] = wmma_bf16(a0, b3, acc[0][3]);
      acc[1][3] = wmma_bf16(a1, b3, acc[1][3]);
    }
#pragma unroll
    for (int nc = 0; nc < 4; nc++) {
      int ncol = n0 + nc * 16 + mlane;
      float ob = o_b[ncol];
#pragma unroll
      for (int tile = 0; tile < 2; tile++) {
#pragma unroll
        for (int r = 0; r < 8; r++) {
          int gm = m0 + tile * 16 + r + 8 * hsel;
          size_t off = (size_t)gm * HID + ncol;
          y[off] = x[off] + acc[tile][nc][r] + ob;
        }
      }
    }
  }
}

// ------------------------------------------------------- FiLM-gated finalize
__global__ __launch_bounds__(256) void finalize_kernel(
    const float* __restrict__ y, const int* __restrict__ nat, const int* __restrict__ nmask,
    const float* __restrict__ rb_tab, const float* __restrict__ pr_g, const float* __restrict__ pr_b,
    const float* __restrict__ r_scale, const float* __restrict__ b_scale,
    float* __restrict__ out) {
  __shared__ float red[16];
  const int row = blockIdx.x, tid = threadIdx.x;
  f32x4 v = ((const f32x4*)(y + (size_t)row * HID))[tid];
  float s = v.x + v.y + v.z + v.w;
  float s2 = v.x * v.x + v.y * v.y + v.z * v.z + v.w * v.w;
#pragma unroll
  for (int msk = 16; msk >= 1; msk >>= 1) { s += __shfl_xor(s, msk); s2 += __shfl_xor(s2, msk); }
  int w = tid >> 5;
  if ((tid & 31) == 0) { red[w] = s; red[8 + w] = s2; }
  __syncthreads();
  float ts = 0.f, ts2 = 0.f;
#pragma unroll
  for (int i = 0; i < 8; i++) { ts += red[i]; ts2 += red[8 + i]; }
  float mean = ts * (1.0f / HID);
  float inv = rsqrtf(ts2 * (1.0f / HID) - mean * mean + 1e-5f);
  int id = (nat[row] + 1) * (nmask[row] == 1 ? 1 : 0);
  const float* rrow = rb_tab + (size_t)id * (2 * HID);
  float rs = r_scale[0], bs = b_scale[0];
  float vv[4] = {v.x, v.y, v.z, v.w};
  float oo[4];
#pragma unroll
  for (int j = 0; j < 4; j++) {
    int col = tid * 4 + j;
    float ln = (vv[j] - mean) * inv * pr_g[col] + pr_b[col];
    oo[j] = vv[j] + ln * tanh_f(rrow[col]) * rs + rrow[HID + col] * bs;
  }
  f32x4 o; o.x = oo[0]; o.y = oo[1]; o.z = oo[2]; o.w = oo[3];
  ((f32x4*)(out + (size_t)row * HID))[tid] = o;
}

// ----------------------------------------------------------------- launcher
extern "C" void kernel_launch(void* const* d_in, const int* in_sizes, int n_in,
                              void* d_out, int out_size, void* d_ws, size_t ws_size,
                              hipStream_t stream) {
  const float* x         = (const float*)d_in[0];
  const int*   interval  = (const int*)d_in[1];
  const int*   nat       = (const int*)d_in[3];
  const int*   nmask     = (const int*)d_in[4];
  const float* ln_g      = (const float*)d_in[5];
  const float* ln_b      = (const float*)d_in[6];
  const float* uvqk      = (const float*)d_in[7];
  const float* o_w       = (const float*)d_in[8];
  const float* o_b       = (const float*)d_in[9];
  const float* pos_w     = (const float*)d_in[10];
  const float* ts_w      = (const float*)d_in[11];
  const float* act_tab   = (const float*)d_in[12];
  const float* film_ln_g = (const float*)d_in[13];
  const float* film_ln_b = (const float*)d_in[14];
  const float* film_w    = (const float*)d_in[15];
  const float* film_b    = (const float*)d_in[16];
  const float* pr_g      = (const float*)d_in[17];
  const float* pr_b      = (const float*)d_in[18];
  const float* r_scale   = (const float*)d_in[19];
  const float* b_scale   = (const float*)d_in[20];

  char* p = (char*)d_ws;
  auto carve = [&](size_t bytes) { void* r = (void*)p; p += (bytes + 255) & ~(size_t)255; return r; };
  unsigned short* uvqkT = (unsigned short*)carve((size_t)DPROJ * HID * 2);
  unsigned short* owT   = (unsigned short*)carve((size_t)HID * 512 * 2);
  float*          rbTab = (float*)carve((size_t)4 * 2 * HID * 4);
  float*          Ubuf  = (float*)carve((size_t)BATCH * NHEAD * S_LEN * LDIM * 4);
  unsigned short* Vt    = (unsigned short*)carve((size_t)BATCH * NHEAD * LDIM * S_LEN * 2);
  unsigned short* Qb    = (unsigned short*)carve((size_t)BATCH * NHEAD * S_LEN * ADIM * 2);
  unsigned short* Kb    = (unsigned short*)carve((size_t)BATCH * NHEAD * S_LEN * ADIM * 2);
  unsigned short* ud    = (unsigned short*)carve((size_t)BATCH * S_LEN * NHEAD * LDIM * 2);
  float*          ybuf  = (float*)carve((size_t)BATCH * S_LEN * HID * 4);

  transpose_bf16_kernel<<<(HID * DPROJ + 255) / 256, 256, 0, stream>>>(uvqk, uvqkT, HID, DPROJ);
  transpose_bf16_kernel<<<(512 * HID + 255) / 256, 256, 0, stream>>>(o_w, owT, 512, HID);
  film_table_kernel<<<4, 256, 0, stream>>>(act_tab, film_ln_g, film_ln_b, film_w, film_b, rbTab);
  ln_proj_kernel<<<BATCH * S_LEN / 32, 256, 0, stream>>>(x, ln_g, ln_b, uvqkT, Ubuf, Vt, Qb, Kb);
  attn_kernel<<<dim3(S_LEN / 16, BATCH), 256, 0, stream>>>(Qb, Kb, Vt, Ubuf, interval, ts_w, pos_w, ud);
  out_gemm_kernel<<<BATCH * S_LEN / 32, 256, 0, stream>>>(ud, owT, x, o_b, ybuf);
  finalize_kernel<<<BATCH * S_LEN, 256, 0, stream>>>(ybuf, nat, nmask, rbTab, pr_g, pr_b,
                                                     r_scale, b_scale, (float*)d_out);
}
